// MultiHeadSelfAttention_43971875176763
// MI455X (gfx1250) — matrix operand order repaired and checked by reading, not compiled, not run on hardware
//
#include <hip/hip_runtime.h>

typedef __attribute__((ext_vector_type(16))) __bf16 v16bf;
typedef __attribute__((ext_vector_type(8)))  __bf16 v8bf;
typedef __attribute__((ext_vector_type(4)))  __bf16 v4bf;
typedef __attribute__((ext_vector_type(8)))  float  v8f;

#define SEQ    2048
#define DMODEL 1024
#define NHEAD  16
#define DHEAD  64
#define BATCH  2
#define MROWS  (BATCH * SEQ)   // 4096

static __device__ __forceinline__ v8f wmma_bf16(v16bf a, v16bf b, v8f c) {
  // D = A(16x32 bf16) * B(32x16 bf16) + C(16x16 f32)
  return __builtin_amdgcn_wmma_f32_16x16x32_bf16(false, a, false, b, (short)0, c,
                                                 false, false);
}

// A-fragment (16x32, row-major bf16 source), contiguous K order (matches B):
// lanes 0-15: elems 0..15 -> K 0..15
// lanes 16-31: elems 0..15 -> K 16..31
static __device__ __forceinline__ v16bf load_a_frag(const __bf16* row, int k, bool hi) {
  const v8bf c0 = *(const v8bf*)(row + k + (hi ? 16 : 0));
  const v8bf c1 = *(const v8bf*)(row + k + 8 + (hi ? 16 : 0));
  v16bf a;
#pragma unroll
  for (int i = 0; i < 8; ++i) { a[i] = c0[i]; a[8 + i] = c1[i]; }
  return a;
}

// ---------------------------------------------------------------------------
// fp32 -> bf16 bulk convert (x)
// ---------------------------------------------------------------------------
__global__ __launch_bounds__(256) void cvt_bf16_kernel(const float* __restrict__ s,
                                                       __bf16* __restrict__ d, int n) {
  int i = (blockIdx.x * 256 + threadIdx.x) * 4;
  if (i + 3 < n) {
    float4 v = *(const float4*)(s + i);
    v4bf o = {(__bf16)v.x, (__bf16)v.y, (__bf16)v.z, (__bf16)v.w};
    *(v4bf*)(d + i) = o;
  }
}

// ---------------------------------------------------------------------------
// weight transpose + convert: src K x N (fp32, row-major) -> dst N x K (bf16)
// ---------------------------------------------------------------------------
__global__ __launch_bounds__(256) void transpose_cvt_kernel(const float* __restrict__ src,
                                                            __bf16* __restrict__ dst,
                                                            int K, int N) {
  __shared__ float tile[64][65];
  const int k0 = blockIdx.y * 64, n0 = blockIdx.x * 64;
  for (int i = threadIdx.x; i < 64 * 64; i += 256) {
    int r = i >> 6, c = i & 63;
    tile[r][c] = src[(size_t)(k0 + r) * N + n0 + c];
  }
  __syncthreads();
  for (int i = threadIdx.x; i < 64 * 64; i += 256) {
    int r = i >> 6, c = i & 63;
    dst[(size_t)(n0 + r) * K + k0 + c] = (__bf16)tile[c][r];
  }
}

// ---------------------------------------------------------------------------
// C[M,N] = A[M,K] * Wt[N,K]^T + bias   (bf16 WMMA, f32 accumulate)
// Block = 256 threads = 8 waves -> 64 x 128 output tile.
// wave w: mt = w&3 (16 rows), nt = w>>2 (64 cols -> four 16x16 accumulators,
// A-fragment reused 4x per K-step).
// TRANS_OUT stores C^T (N x M) with per-lane contiguous 8x bf16 stores.
// ---------------------------------------------------------------------------
template <bool TRANS_OUT, typename OutT>
__global__ __launch_bounds__(256) void gemm_bias_kernel(
    const __bf16* __restrict__ A,    // M x K
    const __bf16* __restrict__ Wt,   // N x K (pre-transposed weights)
    const float*  __restrict__ bias, // N
    OutT* __restrict__ C,            // M x N  (N x M if TRANS_OUT)
    int M, int N, int K) {
  const int tid = threadIdx.x;
  const int w = tid >> 5, l = tid & 31;
  const int lo = l & 15;
  const bool hi = l >= 16;
  const int mt = w & 3, nt = w >> 2;
  const int mBase = blockIdx.y * 64 + mt * 16;
  const int nBase = blockIdx.x * 128 + nt * 64;

  const __bf16* arow = A + (size_t)(mBase + lo) * K;
  const __bf16* brow[4];
#pragma unroll
  for (int j = 0; j < 4; ++j)
    brow[j] = Wt + (size_t)(nBase + j * 16 + lo) * K + (hi ? 16 : 0);

  v8f acc[4] = {};
#pragma unroll 2
  for (int k = 0; k < K; k += 32) {
    v16bf a = load_a_frag(arow, k, hi);
#pragma unroll
    for (int j = 0; j < 4; ++j) {
      v16bf b = *(const v16bf*)(brow[j] + k);
      acc[j] = wmma_bf16(a, b, acc[j]);
    }
  }

#pragma unroll
  for (int j = 0; j < 4; ++j) {
    const int n = nBase + j * 16 + lo;
    const float bi = bias[n];
    if constexpr (TRANS_OUT) {
      v8bf o;
#pragma unroll
      for (int r = 0; r < 8; ++r) o[r] = (__bf16)(acc[j][r] + bi);
      *(v8bf*)(C + (size_t)n * M + mBase + (hi ? 8 : 0)) = o;
    } else {
#pragma unroll
      for (int r = 0; r < 8; ++r) {
        const int m = mBase + r + (hi ? 8 : 0);
        C[(size_t)m * N + n] = (OutT)(acc[j][r] + bi);
      }
    }
  }
}

// ---------------------------------------------------------------------------
// Attention: one block per (b, h, 16-query tile). 256 threads = 8 waves.
// LDS: 16 x SEQ f32 scores (128 KB) + 16 x SEQ bf16 probs (64 KB) + stats.
// ---------------------------------------------------------------------------
__global__ __launch_bounds__(256) void attn_kernel(
    const __bf16* __restrict__ Qb,   // [B*SEQ][DMODEL]
    const __bf16* __restrict__ Kb,   // [B*SEQ][DMODEL]
    const __bf16* __restrict__ Vt,   // [DMODEL][B*SEQ]  (transposed V)
    const int*    __restrict__ mask, // [B][1][SEQ][SEQ]
    __bf16* __restrict__ Ob) {       // [B*SEQ][DMODEL]
  extern __shared__ float smem[];
  float*  sc    = smem;                    // 16 * SEQ f32 scores
  float*  red   = smem + 16 * SEQ;         // 256 reduction scratch
  float*  rstat = red + 256;               // 16 row stats (max, then 1/sum)
  __bf16* pb    = (__bf16*)(rstat + 16);   // 16 * SEQ bf16 probabilities

  const int b   = blockIdx.z;
  const int h   = blockIdx.y;
  const int q0  = blockIdx.x * 16;
  const int tid = threadIdx.x;
  const int w = tid >> 5, l = tid & 31;
  const int lo = l & 15;
  const bool hi = l >= 16;
  const int bm = b * SEQ;

  // ---- scores = Q . K^T : A = Q rows (loaded once), B = K rows ----
  const __bf16* qrow = Qb + (size_t)(bm + q0 + lo) * DMODEL + h * DHEAD;
  const v16bf aq0 = load_a_frag(qrow, 0, hi);
  const v16bf aq1 = load_a_frag(qrow, 32, hi);

  for (int t = w; t < SEQ / 16; t += 8) {
    const int k0 = t * 16;
    const __bf16* krow =
        Kb + (size_t)(bm + k0 + lo) * DMODEL + h * DHEAD + (hi ? 16 : 0);
    v16bf b0 = *(const v16bf*)(krow);        // d = 0..31
    v16bf b1 = *(const v16bf*)(krow + 32);   // d = 32..63
    v8f acc = {};
    acc = wmma_bf16(aq0, b0, acc);
    acc = wmma_bf16(aq1, b1, acc);
#pragma unroll
    for (int r = 0; r < 8; ++r)
      sc[(r + (hi ? 8 : 0)) * SEQ + k0 + lo] = acc[r];
  }
  __syncthreads();

  // ---- mask + scale + softmax; 16 threads per row, interleaved float4/int4
  //      so each iteration reads 256 B contiguous per row (coalesced) ----
  const int row = tid >> 4, ch = tid & 15;
  float4* srow4 = (float4*)(sc + row * SEQ);
  const int4* mrow4 =
      (const int4*)(mask + ((size_t)b * SEQ + (q0 + row)) * SEQ);
  float mx = -3.0e38f;
  for (int i = 0; i < SEQ / 64; ++i) {
    const int j = ch + i * 16;               // float4 index
    const int4 mm = mrow4[j];
    float4 sv = srow4[j];
    sv.x = (mm.x == 0) ? -1.0e9f : sv.x * 0.125f;
    sv.y = (mm.y == 0) ? -1.0e9f : sv.y * 0.125f;
    sv.z = (mm.z == 0) ? -1.0e9f : sv.z * 0.125f;
    sv.w = (mm.w == 0) ? -1.0e9f : sv.w * 0.125f;
    srow4[j] = sv;
    mx = fmaxf(mx, fmaxf(fmaxf(sv.x, sv.y), fmaxf(sv.z, sv.w)));
  }
  red[tid] = mx;
  __syncthreads();
  if (ch == 0) {
    float m = red[row * 16];
#pragma unroll
    for (int i = 1; i < 16; ++i) m = fmaxf(m, red[row * 16 + i]);
    rstat[row] = m;
  }
  __syncthreads();
  const float rmax = rstat[row];
  v4bf* prow4 = (v4bf*)(pb + row * SEQ);
  float s = 0.f;
  for (int i = 0; i < SEQ / 64; ++i) {
    const int j = ch + i * 16;
    const float4 sv = srow4[j];
    const float e0 = __expf(sv.x - rmax);
    const float e1 = __expf(sv.y - rmax);
    const float e2 = __expf(sv.z - rmax);
    const float e3 = __expf(sv.w - rmax);
    s += (e0 + e1) + (e2 + e3);
    v4bf p = {(__bf16)e0, (__bf16)e1, (__bf16)e2, (__bf16)e3};
    prow4[j] = p;                            // bf16 probs for the PV WMMAs
  }
  red[tid] = s;
  __syncthreads();
  if (ch == 0) {
    float t = 0.f;
#pragma unroll
    for (int i = 0; i < 16; ++i) t += red[row * 16 + i];
    rstat[row] = 1.0f / t;
  }
  __syncthreads();

  // ---- out = P . V : split-K across wave pairs, 4 d-tiles of 16.
  //      A-fragments are bf16 straight from LDS (no converts in loop). ----
  const int nt = w & 3, kh = w >> 2;
  v8f acc = {};
  const __bf16* vrow = Vt + (size_t)(h * DHEAD + nt * 16 + lo) * MROWS + bm +
                       kh * 1024 + (hi ? 16 : 0);
  const __bf16* prow = pb + lo * SEQ + kh * 1024;
#pragma unroll 2
  for (int k = 0; k < 1024; k += 32) {
    v16bf a  = load_a_frag(prow, k, hi);
    v16bf bf = *(const v16bf*)(vrow + k);
    acc = wmma_bf16(a, bf, acc);
  }
  __syncthreads();
  if (kh == 1) {
    float* dst = sc + (nt * 32 + l) * 8;     // sc region is free now
#pragma unroll
    for (int r = 0; r < 8; ++r) dst[r] = acc[r];
  }
  __syncthreads();
  if (kh == 0) {
    const float* src = sc + (nt * 32 + l) * 8;
#pragma unroll
    for (int r = 0; r < 8; ++r) {
      const int m = r + (hi ? 8 : 0);
      const float o = (acc[r] + src[r]) * rstat[m];
      Ob[(size_t)(bm + q0 + m) * DMODEL + h * DHEAD + nt * 16 + lo] = (__bf16)o;
    }
  }
}

// ---------------------------------------------------------------------------
extern "C" void kernel_launch(void* const* d_in, const int* in_sizes, int n_in,
                              void* d_out, int out_size, void* d_ws, size_t ws_size,
                              hipStream_t stream) {
  const float* x    = (const float*)d_in[0];
  const int*   mask = (const int*)d_in[1];
  const float* w_q  = (const float*)d_in[2];
  const float* b_q  = (const float*)d_in[3];
  const float* w_k  = (const float*)d_in[4];
  const float* b_k  = (const float*)d_in[5];
  const float* w_v  = (const float*)d_in[6];
  const float* b_v  = (const float*)d_in[7];
  const float* w_o  = (const float*)d_in[8];
  const float* b_o  = (const float*)d_in[9];
  float* out = (float*)d_out;

  char* ws = (char*)d_ws;
  __bf16* xb  = (__bf16*)(ws);                        // 8 MB : x bf16
  __bf16* wqt = (__bf16*)(ws + (size_t)8  * 1048576); // 2 MB each, transposed
  __bf16* wkt = (__bf16*)(ws + (size_t)10 * 1048576);
  __bf16* wvt = (__bf16*)(ws + (size_t)12 * 1048576);
  __bf16* wot = (__bf16*)(ws + (size_t)14 * 1048576);
  __bf16* Qb  = (__bf16*)(ws + (size_t)16 * 1048576); // 8 MB
  __bf16* Kb  = (__bf16*)(ws + (size_t)24 * 1048576); // 8 MB
  __bf16* Vt  = (__bf16*)(ws + (size_t)32 * 1048576); // 8 MB (DMODEL x MROWS)
  __bf16* Ob  = (__bf16*)(ws + (size_t)40 * 1048576); // 8 MB

  // 1) convert inputs to bf16 (weights transposed to N x K)
  cvt_bf16_kernel<<<MROWS * DMODEL / 1024, 256, 0, stream>>>(x, xb, MROWS * DMODEL);
  dim3 tgrid(DMODEL / 64, DMODEL / 64);
  transpose_cvt_kernel<<<tgrid, 256, 0, stream>>>(w_q, wqt, DMODEL, DMODEL);
  transpose_cvt_kernel<<<tgrid, 256, 0, stream>>>(w_k, wkt, DMODEL, DMODEL);
  transpose_cvt_kernel<<<tgrid, 256, 0, stream>>>(w_v, wvt, DMODEL, DMODEL);
  transpose_cvt_kernel<<<tgrid, 256, 0, stream>>>(w_o, wot, DMODEL, DMODEL);

  // 2) Q/K/V projections (V written transposed for PV-matmul locality)
  dim3 ggrid(DMODEL / 128, MROWS / 64);
  gemm_bias_kernel<false, __bf16><<<ggrid, 256, 0, stream>>>(xb, wqt, b_q, Qb,
                                                             MROWS, DMODEL, DMODEL);
  gemm_bias_kernel<false, __bf16><<<ggrid, 256, 0, stream>>>(xb, wkt, b_k, Kb,
                                                             MROWS, DMODEL, DMODEL);
  gemm_bias_kernel<true, __bf16><<<ggrid, 256, 0, stream>>>(xb, wvt, b_v, Vt,
                                                            MROWS, DMODEL, DMODEL);

  // 3) attention (scores+softmax+PV) — ~193 KB dynamic LDS per block
  const size_t smem =
      (size_t)(16 * SEQ + 256 + 16) * sizeof(float) + (size_t)16 * SEQ * 2;
  hipFuncSetAttribute((const void*)attn_kernel,
                      hipFuncAttributeMaxDynamicSharedMemorySize, (int)smem);
  dim3 agrid(SEQ / 16, NHEAD, BATCH);
  attn_kernel<<<agrid, 256, smem, stream>>>(Qb, Kb, Vt, mask, Ob);

  // 4) output projection -> fp32 d_out
  gemm_bias_kernel<false, float><<<ggrid, 256, 0, stream>>>(Ob, wot, b_o, out,
                                                            MROWS, DMODEL, DMODEL);
}
